// TransformerEncoderDecoder_65833258713253
// MI455X (gfx1250) — matrix operand order repaired and checked by reading, not compiled, not run on hardware
//
#include <hip/hip_runtime.h>
#include <hip/hip_bf16.h>
#include <math.h>

// ---------------- CDNA5 WMMA plumbing ----------------
typedef __attribute__((ext_vector_type(16))) _Float16 v16h;
typedef __attribute__((ext_vector_type(8)))  _Float16 v8h;
typedef __attribute__((ext_vector_type(8)))  float    v8f;

__device__ __forceinline__ v8f wmma_f16(v16h a, v16h b, v8f c) {
    // D = A(16x32 f16) * B(32x16 f16) + C(16x16 f32)
    return __builtin_amdgcn_wmma_f32_16x16x32_f16(false, a, false, b, (short)0, c, false, false);
}
__device__ __forceinline__ v16h cat8(v8h lo, v8h hi) {
    return __builtin_shufflevector(lo, hi, 0,1,2,3,4,5,6,7,8,9,10,11,12,13,14,15);
}
// Load 8 contiguous f32, scale, convert to packed f16
__device__ __forceinline__ v8h cvt8(const float* __restrict__ p, float s) {
    float4 x = *(const float4*)p;
    float4 y = *(const float4*)(p + 4);
    v8h r;
    r[0]=(_Float16)(x.x*s); r[1]=(_Float16)(x.y*s); r[2]=(_Float16)(x.z*s); r[3]=(_Float16)(x.w*s);
    r[4]=(_Float16)(y.x*s); r[5]=(_Float16)(y.y*s); r[6]=(_Float16)(y.z*s); r[7]=(_Float16)(y.w*s);
    return r;
}

// ---- Pure-VALU 16-lane butterfly reductions via DPP (no ds_bpermute, no DS waits) ----
// XOR masks {1,2,7,15}: quad_perm(1,0,3,2)=0xB1, quad_perm(2,3,0,1)=0x4E,
// row_half_mirror=0x141 (lane^7), row_mirror=0x140 (lane^15). Linearly independent
// basis -> every lane accumulates all 16 lanes of its half-wave.
template<int CTRL>
__device__ __forceinline__ float dppf(float v) {
    return __int_as_float(__builtin_amdgcn_mov_dpp(__float_as_int(v), CTRL, 0xF, 0xF, true));
}
__device__ __forceinline__ float red_max16(float v) {
    v = fmaxf(v, dppf<0xB1>(v));
    v = fmaxf(v, dppf<0x4E>(v));
    v = fmaxf(v, dppf<0x141>(v));
    v = fmaxf(v, dppf<0x140>(v));
    return v;
}
__device__ __forceinline__ float red_sum16(float v) {
    v += dppf<0xB1>(v);
    v += dppf<0x4E>(v);
    v += dppf<0x141>(v);
    v += dppf<0x140>(v);
    return v;
}
__device__ __forceinline__ float red_sum32(float v) {
    v = red_sum16(v);
    v += __shfl_xor(v, 16, 32);   // single cross-half step
    return v;
}

// ---------------- LayerNorm: one wave32 per row (float4 vectorized) ----------------
__global__ __launch_bounds__(256)
void layernorm_k(const float* __restrict__ x, const float* __restrict__ w,
                 const float* __restrict__ b, float* __restrict__ y,
                 int rows, int cols)
{
    const int wave = threadIdx.x >> 5;
    const int lane = threadIdx.x & 31;
    const int row  = blockIdx.x * 8 + wave;
    if (row >= rows) return;
    const float* xr = x + (size_t)row * cols;
    float s = 0.f, s2 = 0.f;
    for (int c = lane * 4; c < cols; c += 128) {
        float4 v = *(const float4*)(xr + c);
        s  += v.x + v.y + v.z + v.w;
        s2 += v.x*v.x + v.y*v.y + v.z*v.z + v.w*v.w;
    }
    s  = red_sum32(s);
    s2 = red_sum32(s2);
    const float mean = s / cols;
    const float var  = s2 / cols - mean * mean;
    const float rstd = rsqrtf(var + 1e-5f);
    float* yr = y + (size_t)row * cols;
    for (int c = lane * 4; c < cols; c += 128) {
        float4 v = *(const float4*)(xr + c);
        float4 ww = *(const float4*)(w + c);
        float4 bb = *(const float4*)(b + c);
        float4 o;
        o.x = (v.x - mean) * rstd * ww.x + bb.x;
        o.y = (v.y - mean) * rstd * ww.y + bb.y;
        o.z = (v.z - mean) * rstd * ww.z + bb.z;
        o.w = (v.w - mean) * rstd * ww.w + bb.w;
        *(float4*)(yr + c) = o;
    }
}

// ---------------- GEGLU: out = a * gelu_exact(g), float4 vectorized ----------------
__global__ __launch_bounds__(256)
void geglu_k(const float* __restrict__ h, float* __restrict__ g, int rows)
{
    const int idx4 = blockIdx.x * blockDim.x + threadIdx.x;   // one float4 per thread
    const int total4 = rows * 1024;
    if (idx4 >= total4) return;
    const int r = idx4 >> 10, c = (idx4 & 1023) * 4;
    float4 a  = *(const float4*)(h + (size_t)r * 8192 + c);
    float4 gg = *(const float4*)(h + (size_t)r * 8192 + 4096 + c);
    float4 o;
    o.x = a.x * (0.5f * gg.x * (1.f + erff(gg.x * 0.70710678118f)));
    o.y = a.y * (0.5f * gg.y * (1.f + erff(gg.y * 0.70710678118f)));
    o.z = a.z * (0.5f * gg.z * (1.f + erff(gg.z * 0.70710678118f)));
    o.w = a.w * (0.5f * gg.w * (1.f + erff(gg.w * 0.70710678118f)));
    *(float4*)(g + (size_t)r * 4096 + c) = o;
}

// ---------------- GEMM: C[M,N] = A[M,K] @ W[K,N] (+bias) (+residual) ----------------
// Workgroup tile 64x128, K-step 64, software-pipelined: global loads for tile k+1
// are issued while each wave's 8 WMMAs of tile k execute. 8 waves (4Mx2N groups); each wave
// owns four 16x16 accumulators. A staged [m][k]; W staged transposed [n][k] so every
// fragment is a pair of contiguous ds_load_b128.
template<bool BIAS, bool RES>
__global__ __launch_bounds__(256)
void gemm_k(const float* __restrict__ A, const float* __restrict__ W,
            const float* __restrict__ bias, const float* __restrict__ R,
            float* __restrict__ C, int M, int N, int K)
{
    __shared__ _Float16 As[64][72];    // [m][k]  row stride 144B (16B-aligned)
    __shared__ _Float16 Bt[128][72];   // [n][k]  transposed

    const int tid  = threadIdx.x;
    const int lane = tid & 31;
    const int wave = tid >> 5;
    const int wm   = wave & 3;        // M subtile 0..3
    const int wn4  = wave >> 2;       // N group 0..1 (4 subtiles each)
    const int hf   = lane >> 4;
    const int l16  = lane & 15;
    const int m0   = blockIdx.y * 64;
    const int n0   = blockIdx.x * 128;

    // Per-thread staging coordinates
    const int ar = tid >> 2,  ac = (tid & 3) << 4;   // A: 16 contiguous f32
    const int bkr = tid >> 2, bc = (tid & 3) << 5;   // W: 32 contiguous f32

    v8f acc0 = {}, acc1 = {}, acc2 = {}, acc3 = {};
    v8h ra0, ra1, rb0, rb1, rb2, rb3;

    // Prologue: fetch tile 0
    {
        const float* srcA = A + (size_t)(m0 + ar) * K + ac;
        ra0 = cvt8(srcA, 1.f); ra1 = cvt8(srcA + 8, 1.f);
        const float* srcB = W + (size_t)bkr * N + n0 + bc;
        rb0 = cvt8(srcB, 1.f);      rb1 = cvt8(srcB + 8, 1.f);
        rb2 = cvt8(srcB + 16, 1.f); rb3 = cvt8(srcB + 24, 1.f);
    }

    for (int k0 = 0; k0 < K; k0 += 64) {
        // Commit staged registers to LDS
        *(v8h*)&As[ar][ac]     = ra0;
        *(v8h*)&As[ar][ac + 8] = ra1;
        #pragma unroll
        for (int j = 0; j < 8; ++j) {
            Bt[bc +  0 + j][bkr] = rb0[j];
            Bt[bc +  8 + j][bkr] = rb1[j];
            Bt[bc + 16 + j][bkr] = rb2[j];
            Bt[bc + 24 + j][bkr] = rb3[j];
        }
        __syncthreads();

        // Issue next tile's global loads; they overlap the WMMA block below
        if (k0 + 64 < K) {
            const float* srcA = A + (size_t)(m0 + ar) * K + k0 + 64 + ac;
            ra0 = cvt8(srcA, 1.f); ra1 = cvt8(srcA + 8, 1.f);
            const float* srcB = W + (size_t)(k0 + 64 + bkr) * N + n0 + bc;
            rb0 = cvt8(srcB, 1.f);      rb1 = cvt8(srcB + 8, 1.f);
            rb2 = cvt8(srcB + 16, 1.f); rb3 = cvt8(srcB + 24, 1.f);
        }

        #pragma unroll
        for (int kc = 0; kc < 64; kc += 32) {
            const int arow = wm * 16 + l16;
            v16h af = cat8(*(const v8h*)&As[arow][kc + hf * 16],
                           *(const v8h*)&As[arow][kc + 8 + hf * 16]);
            const int kb = kc + hf * 16;
            {
                v16h bf = cat8(*(const v8h*)&Bt[wn4*64 +  0 + l16][kb],
                               *(const v8h*)&Bt[wn4*64 +  0 + l16][kb + 8]);
                acc0 = wmma_f16(af, bf, acc0);
            }
            {
                v16h bf = cat8(*(const v8h*)&Bt[wn4*64 + 16 + l16][kb],
                               *(const v8h*)&Bt[wn4*64 + 16 + l16][kb + 8]);
                acc1 = wmma_f16(af, bf, acc1);
            }
            {
                v16h bf = cat8(*(const v8h*)&Bt[wn4*64 + 32 + l16][kb],
                               *(const v8h*)&Bt[wn4*64 + 32 + l16][kb + 8]);
                acc2 = wmma_f16(af, bf, acc2);
            }
            {
                v16h bf = cat8(*(const v8h*)&Bt[wn4*64 + 48 + l16][kb],
                               *(const v8h*)&Bt[wn4*64 + 48 + l16][kb + 8]);
                acc3 = wmma_f16(af, bf, acc3);
            }
        }
        __syncthreads();
    }

    // Epilogue (C layout: N = lane%16, VGPR r -> M = r + 8*half)
    v8f accs[4] = { acc0, acc1, acc2, acc3 };
    #pragma unroll
    for (int s = 0; s < 4; ++s) {
        const int n = n0 + wn4 * 64 + s * 16 + l16;
        const float bv = BIAS ? bias[n] : 0.f;
        #pragma unroll
        for (int r = 0; r < 8; ++r) {
            const int m = m0 + wm * 16 + r + hf * 8;
            float v = accs[s][r] + bv;
            if (RES) v += R[(size_t)m * N + n];
            C[(size_t)m * N + n] = v;
        }
    }
}

// ---------------- Flash attention (dh = 64) ----------------
// Grid: (Nq/64, H, B), block 128 (4 waves, 16 query rows each).
__global__ __launch_bounds__(128)
void flash_attn_k(const float* __restrict__ Q, const float* __restrict__ KV,
                  float* __restrict__ O, const unsigned char* __restrict__ mask,
                  int Nq, int Nk, int qld, int kvld, int voff, float scale)
{
    __shared__ _Float16 Ks[32][72];      // [kv][dh]  (stride 144B)
    __shared__ _Float16 Vt[64][40];      // [dh][kv]  transposed (stride 80B)
    __shared__ _Float16 Ps[4][16][40];   // per-wave P tile 16x32

    const int b = blockIdx.z, h = blockIdx.y;
    const int tid  = threadIdx.x;
    const int lane = tid & 31;
    const int wave = tid >> 5;
    const int hf   = lane >> 4;
    const int l16  = lane & 15;
    const int qr   = blockIdx.x * 64 + wave * 16;

    const float* Qb = Q  + (size_t)b * Nq * qld  + (size_t)h * 64;
    const float* Kb = KV + (size_t)b * Nk * kvld + (size_t)h * 64;
    const float* Vb = KV + (size_t)b * Nk * kvld + voff + (size_t)h * 64;
    float*       Ob = O  + (size_t)b * Nq * qld  + (size_t)h * 64;

    const int koffA = hf * 16;
    const int koffB = hf * 16;

    // Q fragments loaded once, pre-scaled (dh=64 -> two A fragments)
    const float* qrow = Qb + (size_t)(qr + l16) * qld;
    v16h aq0 = cat8(cvt8(qrow + koffA,      scale), cvt8(qrow + 8 + koffA, scale));
    v16h aq1 = cat8(cvt8(qrow + 32 + koffA, scale), cvt8(qrow + 40 + koffA, scale));

    v8f o0 = {}, o1 = {}, o2 = {}, o3 = {};
    float mrow[8], lrow[8];
    #pragma unroll
    for (int r = 0; r < 8; ++r) { mrow[r] = -1e30f; lrow[r] = 0.f; }

    for (int kv0 = 0; kv0 < Nk; kv0 += 32) {
        // Stage K [kv][dh] vectorized; V transposed [dh][kv]
        {
            const int r = tid >> 2;            // 0..31 key row
            const int c = (tid & 3) << 4;      // 0,16,32,48 dh col
            const float* ksrc = Kb + (size_t)(kv0 + r) * kvld + c;
            const float* vsrc = Vb + (size_t)(kv0 + r) * kvld + c;
            *(v8h*)&Ks[r][c]     = cvt8(ksrc,     1.f);
            *(v8h*)&Ks[r][c + 8] = cvt8(ksrc + 8, 1.f);
            v8h t0 = cvt8(vsrc, 1.f), t1 = cvt8(vsrc + 8, 1.f);
            #pragma unroll
            for (int j = 0; j < 8; ++j) { Vt[c + j][r] = t0[j]; Vt[c + 8 + j][r] = t1[j]; }
        }
        __syncthreads();

        // S = Q * K^T : two 16-col subtiles, K(dh)=64 split in 2 chunks
        v8f s0 = {}, s1 = {};
        {
            v16h bk = cat8(*(const v8h*)&Ks[l16][koffB], *(const v8h*)&Ks[l16][koffB + 8]);
            s0 = wmma_f16(aq0, bk, s0);
            bk = cat8(*(const v8h*)&Ks[l16][32 + koffB], *(const v8h*)&Ks[l16][32 + koffB + 8]);
            s0 = wmma_f16(aq1, bk, s0);
            bk = cat8(*(const v8h*)&Ks[16 + l16][koffB], *(const v8h*)&Ks[16 + l16][koffB + 8]);
            s1 = wmma_f16(aq0, bk, s1);
            bk = cat8(*(const v8h*)&Ks[16 + l16][32 + koffB], *(const v8h*)&Ks[16 + l16][32 + koffB + 8]);
            s1 = wmma_f16(aq1, bk, s1);
        }

        // Key-padding mask: lane owns one key column per subtile
        float madd0 = 0.f, madd1 = 0.f;
        if (mask) {
            madd0 = mask[(size_t)b * Nk + kv0 + l16]      ? 0.f : -1e30f;
            madd1 = mask[(size_t)b * Nk + kv0 + 16 + l16] ? 0.f : -1e30f;
        }

        // Online softmax in C-layout registers (row reductions = DPP butterflies)
        #pragma unroll
        for (int r = 0; r < 8; ++r) {
            const float v0 = s0[r] + madd0;
            const float v1 = s1[r] + madd1;
            const float tmax  = red_max16(fmaxf(v0, v1));
            const float mnew  = fmaxf(mrow[r], tmax);
            const float alpha = __expf(mrow[r] - mnew);
            const float p0 = __expf(v0 - mnew);
            const float p1 = __expf(v1 - mnew);
            lrow[r] = lrow[r] * alpha + red_sum16(p0 + p1);
            mrow[r] = mnew;
            o0[r] *= alpha; o1[r] *= alpha; o2[r] *= alpha; o3[r] *= alpha;
            Ps[wave][r + hf * 8][l16]      = (_Float16)p0;
            Ps[wave][r + hf * 8][16 + l16] = (_Float16)p1;
        }

        // Reshape P (C layout) -> A fragment via per-wave LDS (same-wave DS ops ordered)
        v16h ap = cat8(*(const v8h*)&Ps[wave][l16][koffA],
                       *(const v8h*)&Ps[wave][l16][8 + koffA]);

        // O += P * V, via transposed V tile (contiguous b128 fragment loads)
        {
            v16h bv = cat8(*(const v8h*)&Vt[l16][koffB],      *(const v8h*)&Vt[l16][koffB + 8]);
            o0 = wmma_f16(ap, bv, o0);
            bv = cat8(*(const v8h*)&Vt[16 + l16][koffB],      *(const v8h*)&Vt[16 + l16][koffB + 8]);
            o1 = wmma_f16(ap, bv, o1);
            bv = cat8(*(const v8h*)&Vt[32 + l16][koffB],      *(const v8h*)&Vt[32 + l16][koffB + 8]);
            o2 = wmma_f16(ap, bv, o2);
            bv = cat8(*(const v8h*)&Vt[48 + l16][koffB],      *(const v8h*)&Vt[48 + l16][koffB + 8]);
            o3 = wmma_f16(ap, bv, o3);
        }
        __syncthreads();
    }

    // Normalize and write out
    #pragma unroll
    for (int r = 0; r < 8; ++r) {
        const float inv = 1.f / lrow[r];
        float* orow = Ob + (size_t)(qr + r + hf * 8) * qld;
        orow[l16]      = o0[r] * inv;
        orow[16 + l16] = o1[r] * inv;
        orow[32 + l16] = o2[r] * inv;
        orow[48 + l16] = o3[r] * inv;
    }
}

// ---------------- Host-side pipeline ----------------
extern "C" void kernel_launch(void* const* d_in, const int* in_sizes, int n_in,
                              void* d_out, int out_size, void* d_ws, size_t ws_size,
                              hipStream_t stream) {
    (void)in_sizes; (void)n_in; (void)out_size; (void)ws_size;
    const int E = 4, DL = 2, B = 4, N = 1024, QN = 512, D = 1024, QD = 512;
    const int INNER = 1024, CINNER = 512;
    const int ER = B * N;   // 4096 encoder rows
    const int DR = B * QN;  // 2048 decoder rows

    const float* in_x        = (const float*)d_in[0];
    const float* in_queries  = (const float*)d_in[1];
    const unsigned char* msk = (const unsigned char*)d_in[2];
    const float* enc_ln1_w = (const float*)d_in[3],  *enc_ln1_b = (const float*)d_in[4];
    const float* enc_q_w   = (const float*)d_in[5],  *enc_kv_w  = (const float*)d_in[6];
    const float* enc_out_w = (const float*)d_in[7],  *enc_out_b = (const float*)d_in[8];
    const float* enc_ln2_w = (const float*)d_in[9],  *enc_ln2_b = (const float*)d_in[10];
    const float* enc_ff1_w = (const float*)d_in[11], *enc_ff1_b = (const float*)d_in[12];
    const float* enc_ff2_w = (const float*)d_in[13], *enc_ff2_b = (const float*)d_in[14];
    const float* dcr_lnq_w = (const float*)d_in[15], *dcr_lnq_b = (const float*)d_in[16];
    const float* dcr_lnc_w = (const float*)d_in[17], *dcr_lnc_b = (const float*)d_in[18];
    const float* dcr_q_w   = (const float*)d_in[19], *dcr_kv_w  = (const float*)d_in[20];
    const float* dcr_out_w = (const float*)d_in[21], *dcr_out_b = (const float*)d_in[22];
    const float* dsa_ln_w  = (const float*)d_in[23], *dsa_ln_b  = (const float*)d_in[24];
    const float* dsa_q_w   = (const float*)d_in[25], *dsa_kv_w  = (const float*)d_in[26];
    const float* dsa_out_w = (const float*)d_in[27], *dsa_out_b = (const float*)d_in[28];
    const float* dff_ln_w  = (const float*)d_in[29], *dff_ln_b  = (const float*)d_in[30];
    const float* dff_ff1_w = (const float*)d_in[31], *dff_ff1_b = (const float*)d_in[32];
    const float* dff_ff2_w = (const float*)d_in[33], *dff_ff2_b = (const float*)d_in[34];
    const float* logits_w  = (const float*)d_in[35], *logits_b  = (const float*)d_in[36];

    // Workspace carve-out (floats)
    float* ws = (float*)d_ws;
    size_t off = 0;
    auto alloc = [&](size_t n) { float* p = ws + off; off += n; return p; };
    float* xbuf = alloc((size_t)ER * D);
    float* xn   = alloc((size_t)ER * D);
    float* qb   = alloc((size_t)ER * INNER);
    float* kvb  = alloc((size_t)ER * 2 * INNER);
    float* ob   = alloc((size_t)ER * INNER);
    float* hb   = alloc((size_t)ER * 8 * D);
    float* gb   = alloc((size_t)ER * 4 * D);
    float* qnb  = alloc((size_t)DR * QD);
    float* xd0  = alloc((size_t)DR * D);
    float* xd1  = alloc((size_t)DR * D);

    hipMemcpyAsync(xbuf, in_x, (size_t)ER * D * sizeof(float),
                   hipMemcpyDeviceToDevice, stream);

    auto gemm = [&](const float* A, const float* W, const float* bias, const float* R,
                    float* C, int M, int Nn, int K) {
        dim3 g(Nn / 128, M / 64);
        if (bias && R)      gemm_k<true,  true ><<<g, 256, 0, stream>>>(A, W, bias, R, C, M, Nn, K);
        else if (bias)      gemm_k<true,  false><<<g, 256, 0, stream>>>(A, W, bias, R, C, M, Nn, K);
        else                gemm_k<false, false><<<g, 256, 0, stream>>>(A, W, bias, R, C, M, Nn, K);
    };
    auto ln = [&](const float* X, const float* w, const float* b, float* Y, int rows, int cols) {
        layernorm_k<<<(rows + 7) / 8, 256, 0, stream>>>(X, w, b, Y, rows, cols);
    };
    const float scale = 0.125f; // 64^-0.5

    // ---- Encoder: 4x (prenorm self-attn + prenorm GEGLU FF, residual) ----
    for (int i = 0; i < E; ++i) {
        ln(xbuf, enc_ln1_w + (size_t)i * D, enc_ln1_b + (size_t)i * D, xn, ER, D);
        gemm(xn, enc_q_w  + (size_t)i * D * INNER,     nullptr, nullptr, qb,  ER, INNER,     D);
        gemm(xn, enc_kv_w + (size_t)i * D * 2 * INNER, nullptr, nullptr, kvb, ER, 2 * INNER, D);
        flash_attn_k<<<dim3(N / 64, 16, B), 128, 0, stream>>>(
            qb, kvb, ob, msk, N, N, INNER, 2 * INNER, INNER, scale);
        gemm(ob, enc_out_w + (size_t)i * INNER * D, enc_out_b + (size_t)i * D, xbuf, xbuf, ER, D, INNER);
        ln(xbuf, enc_ln2_w + (size_t)i * D, enc_ln2_b + (size_t)i * D, xn, ER, D);
        gemm(xn, enc_ff1_w + (size_t)i * D * 8 * D, enc_ff1_b + (size_t)i * 8 * D, nullptr, hb, ER, 8 * D, D);
        geglu_k<<<(ER * 1024 + 255) / 256, 256, 0, stream>>>(hb, gb, ER);
        gemm(gb, enc_ff2_w + (size_t)i * 4 * D * D, enc_ff2_b + (size_t)i * D, xbuf, xbuf, ER, D, 4 * D);
    }

    // ---- Decoder: 2x (cross-attn replaces x; self-attn + FF residual) ----
    const float* ctx = xbuf;
    int ctxTok = N;
    float* xd[2] = { xd0, xd1 };
    for (int i = 0; i < DL; ++i) {
        ln(in_queries, dcr_lnq_w + (size_t)i * QD, dcr_lnq_b + (size_t)i * QD, qnb, DR, QD);
        ln(ctx,        dcr_lnc_w + (size_t)i * D,  dcr_lnc_b + (size_t)i * D,  xn, B * ctxTok, D);
        gemm(qnb, dcr_q_w  + (size_t)i * QD * CINNER,    nullptr, nullptr, qb,  DR,         CINNER,     QD);
        gemm(xn,  dcr_kv_w + (size_t)i * D * 2 * CINNER, nullptr, nullptr, kvb, B * ctxTok, 2 * CINNER, D);
        flash_attn_k<<<dim3(QN / 64, 8, B), 128, 0, stream>>>(
            qb, kvb, ob, nullptr, QN, ctxTok, CINNER, 2 * CINNER, CINNER, scale);
        gemm(ob, dcr_out_w + (size_t)i * CINNER * D, dcr_out_b + (size_t)i * D, nullptr, xd[i], DR, D, CINNER);

        ln(xd[i], dsa_ln_w + (size_t)i * D, dsa_ln_b + (size_t)i * D, xn, DR, D);
        gemm(xn, dsa_q_w  + (size_t)i * D * INNER,     nullptr, nullptr, qb,  DR, INNER,     D);
        gemm(xn, dsa_kv_w + (size_t)i * D * 2 * INNER, nullptr, nullptr, kvb, DR, 2 * INNER, D);
        flash_attn_k<<<dim3(QN / 64, 16, B), 128, 0, stream>>>(
            qb, kvb, ob, nullptr, QN, QN, INNER, 2 * INNER, INNER, scale);
        gemm(ob, dsa_out_w + (size_t)i * INNER * D, dsa_out_b + (size_t)i * D, xd[i], xd[i], DR, D, INNER);

        ln(xd[i], dff_ln_w + (size_t)i * D, dff_ln_b + (size_t)i * D, xn, DR, D);
        gemm(xn, dff_ff1_w + (size_t)i * D * 8 * D, dff_ff1_b + (size_t)i * 8 * D, nullptr, hb, DR, 8 * D, D);
        geglu_k<<<(DR * 1024 + 255) / 256, 256, 0, stream>>>(hb, gb, DR);
        gemm(gb, dff_ff2_w + (size_t)i * 4 * D * D, dff_ff2_b + (size_t)i * D, xd[i], xd[i], DR, D, 4 * D);

        ctx = xd[i];
        ctxTok = QN;
    }

    // ---- Logits ----
    gemm(xd[DL - 1], logits_w, logits_b, nullptr, (float*)d_out, DR, 1024, D);
}
